// LateralEI_45569603010670
// MI455X (gfx1250) — matrix operand order repaired and checked by reading, not compiled, not run on hardware
//
#include <hip/hip_runtime.h>

typedef __attribute__((ext_vector_type(16))) _Float16 v16h;
typedef __attribute__((ext_vector_type(8)))  _Float16 v8h;
typedef __attribute__((ext_vector_type(4)))  _Float16 v4h;
typedef __attribute__((ext_vector_type(8)))  float    v8f;
typedef __attribute__((ext_vector_type(4)))  float    v4f;
typedef int i32x4 __attribute__((vector_size(16)));

#define D_LAT 64
#define D_H   128
#define BM    128      // output rows per block (8 waves x 16)
#define BN    32       // j-chunk width (= WMMA K for GEMM2)
#define NSPLIT 4       // j-range split for occupancy (deterministic partials + reduce)

// exp(-d2/(2*sigma^2)) = exp2(C*d2);  C = -log2(e)/(2*sigma^2)
#define C_E (-2.0037431123457824f)   // sigma_e = 0.6
#define C_I (-0.5009357780864456f)   // sigma_i = 1.2
#define K_E 0.8f
#define K_I 1.0f

#if defined(__has_builtin)
#if __has_builtin(__builtin_amdgcn_global_load_async_to_lds_b128)
#define USE_ASYNC 1
#endif
#endif
#ifndef USE_ASYNC
#define USE_ASYNC 0
#endif

#if USE_ASYNC
static __device__ __forceinline__ void async_cp16(const void* g, void* l) {
    // 16B global -> LDS via async DMA path (tracked by ASYNCcnt)
    __builtin_amdgcn_global_load_async_to_lds_b128(
        (__attribute__((address_space(1))) i32x4*)g,
        (__attribute__((address_space(3))) i32x4*)l, 0, 0);
}
#define WAIT_ASYNC(n) asm volatile("s_wait_asynccnt " #n ::: "memory")
#else
#define WAIT_ASYNC(n) do {} while (0)
#endif

static __device__ __forceinline__ v16h make16(v8h lo, v8h hi) {
    return __builtin_shufflevector(lo, hi, 0,1,2,3,4,5,6,7,8,9,10,11,12,13,14,15);
}

// ---- prep: z -> f16 copy + row squared norms ----
__global__ void lei_prep_z(const float* __restrict__ z, _Float16* __restrict__ z16,
                           float* __restrict__ sq, int n) {
    int r = blockIdx.x * blockDim.x + threadIdx.x;
    if (r >= n) return;
    const float* zr = z + (size_t)r * D_LAT;
    _Float16* zo = z16 + (size_t)r * D_LAT;
    float s = 0.f;
    #pragma unroll
    for (int c = 0; c < D_LAT; c += 4) {
        v4f v = *(const v4f*)(zr + c);
        s += v[0]*v[0] + v[1]*v[1] + v[2]*v[2] + v[3]*v[3];
        v4h o; o[0] = (_Float16)v[0]; o[1] = (_Float16)v[1];
               o[2] = (_Float16)v[2]; o[3] = (_Float16)v[3];
        *(v4h*)(zo + c) = o;
    }
    sq[r] = s;
}

// ---- prep: h -> f16, transposed to h16T[n][k] (LDS-tiled transpose) ----
__global__ __launch_bounds__(256) void lei_prep_hT(const float* __restrict__ h,
                                                   _Float16* __restrict__ h16T, int n) {
    __shared__ float tile[32][33];
    const int kb = blockIdx.x * 32, nb = blockIdx.y * 32;
    const int r  = threadIdx.x >> 3;
    const int c4 = (threadIdx.x & 7) * 4;
    v4f v = *(const v4f*)(h + (size_t)(kb + r) * D_H + nb + c4);
    tile[r][c4 + 0] = v[0]; tile[r][c4 + 1] = v[1];
    tile[r][c4 + 2] = v[2]; tile[r][c4 + 3] = v[3];
    __syncthreads();
    v4h o;
    #pragma unroll
    for (int q = 0; q < 4; q++) o[q] = (_Float16)tile[c4 + q][r];
    *(v4h*)(h16T + (size_t)(nb + r) * n + kb + c4) = o;
}

// ---- fused: S = z_i z_j^T -> K(S) -> partial += K h_j ----
__global__ __launch_bounds__(256) void lei_main(const _Float16* __restrict__ z16,
                                                const _Float16* __restrict__ h16T,
                                                const float* __restrict__ sqv,
                                                float* __restrict__ part, int n) {
    __shared__ alignas(64) _Float16 zjb[3][BN][D_LAT];   // z_j chunks (triple buffered), 3x4KB
    __shared__ alignas(64) _Float16 hTb[3][D_H][BN];     // h chunks, transposed, 3x8KB
    __shared__ alignas(64) _Float16 kst[8][16][BN];      // per-wave K-tile staging, 8KB

    const int tid = threadIdx.x;
    const int w   = tid >> 5;
    const int l   = tid & 31;
    const int hs  = l >> 4;
    const int lm  = l & 15;
    const int aoff = 16 * hs;
    const int row_base = blockIdx.x * BM + w * 16;
    const int seg = blockIdx.y;
    const int nseg = n / NSPLIT;
    const int jstart = seg * nseg;
    const int NC = nseg / BN;

    // Loop-invariant A operands (z_i strip, 16x64) in WMMA A layout
    const _Float16* zr = z16 + (size_t)(row_base + lm) * D_LAT;
    v16h a0 = make16(*(const v8h*)(zr + aoff),      *(const v8h*)(zr + 8 + aoff));
    v16h a1 = make16(*(const v8h*)(zr + 32 + aoff), *(const v8h*)(zr + 40 + aoff));

    float sqi[8];
    #pragma unroll
    for (int v = 0; v < 8; v++) sqi[v] = sqv[row_base + v + 8 * hs];

    v8f cz = {};
    v8f acc[8];
    #pragma unroll
    for (int t = 0; t < 8; t++) acc[t] = cz;

    // --- chunk staging: contiguous 16B copies (async DMA when available) ---
    auto stage = [&](int jb, int b) {
        // z chunk: 32 rows x 128B, contiguous 4KB; one 16B segment per thread
        {
            const char* src = (const char*)z16 + (size_t)jb * D_LAT * 2 + tid * 16;
            char* dst = (char*)(&zjb[b][0][0]) + tid * 16;
#if USE_ASYNC
            async_cp16(src, dst);
#else
            *(v8h*)dst = *(const v8h*)src;
#endif
        }
        // h chunk: 128 rows x 64B from pre-transposed h16T; two 16B segments per thread
        #pragma unroll
        for (int q = 0; q < 2; q++) {
            int s = tid * 2 + q;                 // 0..511
            int nr = s >> 2, p = s & 3;
            const char* src = (const char*)h16T + ((size_t)nr * n + jb) * 2 + p * 16;
            char* dst = (char*)(&hTb[b][0][0]) + s * 16;
#if USE_ASYNC
            async_cp16(src, dst);
#else
            *(v8h*)dst = *(const v8h*)src;
#endif
        }
    };

    stage(jstart, 0);   // prologue: chunk 0

    for (int c = 0; c < NC; ++c) {
        const int jb  = jstart + c * BN;
        const int buf = c % 3;

        if (c + 1 < NC) {          // pipeline: copy chunk c+1 while computing chunk c
            stage(jb + BN, (c + 1) % 3);
            WAIT_ASYNC(3);         // in-order ASYNCcnt: <=3 outstanding => chunk c landed
        } else {
            WAIT_ASYNC(0);
        }
        __syncthreads();

        // ---- GEMM1: two 16x16 S tiles, K = 64 ----
        const _Float16* zb0 = &zjb[buf][lm][16 * hs];
        const _Float16* zb1 = &zjb[buf][16 + lm][16 * hs];
        v16h b00 = *(const v16h*)(zb0);
        v16h b01 = *(const v16h*)(zb0 + 32);
        v16h b10 = *(const v16h*)(zb1);
        v16h b11 = *(const v16h*)(zb1 + 32);

        v8f s0 = __builtin_amdgcn_wmma_f32_16x16x32_f16(false, a0, false, b00, (short)0, cz, false, false);
        s0     = __builtin_amdgcn_wmma_f32_16x16x32_f16(false, a1, false, b01, (short)0, s0, false, false);
        v8f s1 = __builtin_amdgcn_wmma_f32_16x16x32_f16(false, a0, false, b10, (short)0, cz, false, false);
        s1     = __builtin_amdgcn_wmma_f32_16x16x32_f16(false, a1, false, b11, (short)0, s1, false, false);

        // ---- epilogue on C/D layout: d2 -> Ke - Ki, stash f16 K-tile to LDS ----
        float sqj0 = sqv[jb + lm];
        float sqj1 = sqv[jb + 16 + lm];
        #pragma unroll
        for (int v = 0; v < 8; v++) {
            float d0 = fmaxf(sqi[v] + sqj0 - 2.0f * s0[v], 0.f);
            float d1 = fmaxf(sqi[v] + sqj1 - 2.0f * s1[v], 0.f);
            float k0 = K_E * __builtin_amdgcn_exp2f(C_E * d0) - K_I * __builtin_amdgcn_exp2f(C_I * d0);
            float k1 = K_E * __builtin_amdgcn_exp2f(C_E * d1) - K_I * __builtin_amdgcn_exp2f(C_I * d1);
            kst[w][v + 8 * hs][lm]      = (_Float16)k0;
            kst[w][v + 8 * hs][16 + lm] = (_Float16)k1;
        }
        __builtin_amdgcn_wave_barrier();   // per-wave LDS is in-order in HW; pin compiler order

        // ---- GEMM2: acc(16x128) += K-tile(16x32) * h-chunk(32x128) ----
        const _Float16* kr = &kst[w][lm][0];
        v16h ak = make16(*(const v8h*)(kr + aoff), *(const v8h*)(kr + 8 + aoff));
        #pragma unroll
        for (int t = 0; t < 8; t++) {
            v16h bh = *(const v16h*)(&hTb[buf][t * 16 + lm][16 * hs]);
            acc[t] = __builtin_amdgcn_wmma_f32_16x16x32_f16(false, ak, false, bh, (short)0, acc[t], false, false);
        }
    }

    float* pout = part + (size_t)seg * n * D_H;
    #pragma unroll
    for (int t = 0; t < 8; t++) {
        #pragma unroll
        for (int v = 0; v < 8; v++) {
            pout[(size_t)(row_base + v + 8 * hs) * D_H + t * 16 + lm] = acc[t][v];
        }
    }
}

// ---- deterministic reduction of j-segment partials ----
__global__ void lei_reduce(const float* __restrict__ part, float* __restrict__ out,
                           int total, size_t stride) {
    int i = blockIdx.x * blockDim.x + threadIdx.x;
    if (i >= total) return;
    float s = 0.f;
    #pragma unroll
    for (int seg = 0; seg < NSPLIT; seg++) s += part[(size_t)seg * stride + i];
    out[i] = s;
}

extern "C" void kernel_launch(void* const* d_in, const int* in_sizes, int n_in,
                              void* d_out, int out_size, void* d_ws, size_t ws_size,
                              hipStream_t stream) {
    const float* z = (const float*)d_in[0];
    const float* h = (const float*)d_in[1];
    float* out = (float*)d_out;
    const int n = in_sizes[0] / D_LAT;   // 8192

    // ws: z16 (n*64 f16) | h16T (n*128 f16, transposed) | sq (n f32) | partials (NSPLIT*n*128 f32)
    char* wsb = (char*)d_ws;
    _Float16* z16  = (_Float16*)wsb;
    _Float16* h16T = (_Float16*)(wsb + (size_t)n * D_LAT * sizeof(_Float16));
    float*    sqv  = (float*)   (wsb + (size_t)n * (D_LAT + D_H) * sizeof(_Float16));
    float*    part = (float*)   (wsb + (size_t)n * (D_LAT + D_H) * sizeof(_Float16)
                                     + (size_t)n * sizeof(float));

    lei_prep_z<<<(n + 255) / 256, 256, 0, stream>>>(z, z16, sqv, n);
    lei_prep_hT<<<dim3(n / 32, D_H / 32), 256, 0, stream>>>(h, h16T, n);
    lei_main<<<dim3(n / BM, NSPLIT), 256, 0, stream>>>(z16, h16T, sqv, part, n);
    int total = n * D_H;
    lei_reduce<<<(total + 255) / 256, 256, 0, stream>>>(part, out, total, (size_t)n * D_H);
}
